// SegTransformerDecoder_70007966924885
// MI455X (gfx1250) — compile-verified, matrix operand order checked by reading, not run on hardware
//
#include <hip/hip_runtime.h>
#include <math.h>

// ---------------------------------------------------------------------------
// SegTransformerDecoder forward for MI455X (gfx1250, wave32, WMMA bf16)
//   B=1, h=w=128 (Q=16384), D=128, G=1, P=8, L=4, N=6 cams, C=128
// GEMM-shaped work (2x conv3x3 implicit GEMM + 3 mid-MLP GEMMs) runs on
// v_wmma_f32_16x16x32_bf16 (f32 accumulate) with 16x64 register blocking
// (one wave -> 4 accumulator tiles, shared fragment loaded once per K-step).
// Gather work (deformable sampling) is VALU with channel-last L2-resident
// features.
// ---------------------------------------------------------------------------

typedef __bf16 bf16;
typedef __attribute__((ext_vector_type(16))) __bf16 v16bf;
typedef __attribute__((ext_vector_type(8)))  float  v8f;

#define HW   16384   // 128*128 BEV pixels
#define DCH  128     // channels
#define NCAM 6
#define PPT  8       // points per query
#define NLVL 4

union VecBF {
  uint4 u[2];
  v16bf v;
};

__device__ __forceinline__ float gelu_exact(float x) {
  return 0.5f * x * (1.0f + erff(x * 0.70710678118654752440f));
}

// ---------------------------------------------------------------------------
// Dense GEMM: out[M,N] = act(A[M,K] @ W[N,K]^T + bias[N])
// A, W bf16 row-major. One wave (32 threads) per 16(M) x 64(N) strip:
// the A fragment is loaded once per K-step and reused by 4 WMMAs.
// Per-lane layout follows ISA 7.12.2:
//   A: M = lane&15, K-groups (k0 + {0,16}) for lanes 0-15, (k0 + {8,24}) for 16-31
//   B: N = lane&15, same K-group split;  C/D: N = lane&15, M = r + 8*(lane>=16)
// ---------------------------------------------------------------------------
__global__ void k_gemm_wmma(const bf16* __restrict__ A, const bf16* __restrict__ W,
                            const float* __restrict__ bias, void* __restrict__ out,
                            int K, int N, int act, int out_bf16) {
  const int lane = threadIdx.x;
  const int half = lane >> 4;
  const int mr   = lane & 15;
  const int n0   = blockIdx.y * 64;
  const bf16* arow = A + (size_t)(blockIdx.x * 16 + mr) * K + half * 8;
  const bf16* brow = W + (size_t)(n0 + mr) * K + half * 8;   // + j*16*K per tile
  v8f acc[4];
#pragma unroll
  for (int j = 0; j < 4; ++j) acc[j] = (v8f){0.f, 0.f, 0.f, 0.f, 0.f, 0.f, 0.f, 0.f};
  for (int k0 = 0; k0 < K; k0 += 32) {
    VecBF a;
    a.u[0] = *(const uint4*)(arow + k0);
    a.u[1] = *(const uint4*)(arow + k0 + 16);
    if (k0 + 32 < K) __builtin_prefetch(arow + k0 + 32, 0, 1);
#pragma unroll
    for (int j = 0; j < 4; ++j) {
      VecBF b;
      const bf16* bj = brow + (size_t)j * 16 * K + k0;
      b.u[0] = *(const uint4*)(bj);
      b.u[1] = *(const uint4*)(bj + 16);
      acc[j] = __builtin_amdgcn_wmma_f32_16x16x32_bf16(false, a.v, false, b.v,
                                                       (short)0, acc[j], false, false);
    }
  }
#pragma unroll
  for (int j = 0; j < 4; ++j) {
    const int n = n0 + j * 16 + mr;
    const float bs = bias[n];
#pragma unroll
    for (int r = 0; r < 8; ++r) {
      const int m = blockIdx.x * 16 + r + half * 8;
      float v = acc[j][r] + bs;
      if (act) v = gelu_exact(v);
      if (out_bf16) ((bf16*)out)[(size_t)m * N + n] = (bf16)v;
      else          ((float*)out)[(size_t)m * N + n] = v;
    }
  }
}

// ---------------------------------------------------------------------------
// Implicit-GEMM 3x3 conv, 128->128, pad 1, on 128x128 image.
//   A = packed weights wpk[o][tap][cin] (K = 9*128 = 1152), bf16
//   B = channel-last input xT[pix][cin], bf16, zero-fill at borders
//   out = fp32 NCHW: y[ch][pix] = conv + bias[ch] + res[ch][pix]
// One wave per 64(ch) x 16(pix) strip: the pixel-patch B fragment is loaded
// once per K-step and reused by 4 WMMAs (weights are tiny and cache-hot).
// grid(1024 pixel tiles, 2 channel strips), block(32).
// ---------------------------------------------------------------------------
__global__ void k_conv3x3_wmma(const bf16* __restrict__ xT, const bf16* __restrict__ wpk,
                               const float* __restrict__ bias, const float* __restrict__ res,
                               float* __restrict__ yout) {
  const int lane = threadIdx.x;
  const int half = lane >> 4;
  const int mr   = lane & 15;
  const int c0   = blockIdx.y * 64;
  const bf16* arow = wpk + (size_t)(c0 + mr) * 1152 + half * 8;  // + j*16*1152 per tile
  const int pix = blockIdx.x * 16 + mr;
  const int py = pix >> 7, px = pix & 127;
  v8f acc[4];
#pragma unroll
  for (int j = 0; j < 4; ++j) acc[j] = (v8f){0.f, 0.f, 0.f, 0.f, 0.f, 0.f, 0.f, 0.f};
  for (int t = 0; t < 9; ++t) {
    const int sy = py + t / 3 - 1;
    const int sx = px + t % 3 - 1;
    const bool ok = (sy >= 0) && (sy < 128) && (sx >= 0) && (sx < 128);
    const bf16* bbase = xT + (size_t)((sy << 7) + sx) * DCH + half * 8;
    const bf16* abase = arow + t * DCH;
#pragma unroll
    for (int kc = 0; kc < 128; kc += 32) {
      VecBF b;
      if (ok) {
        b.u[0] = *(const uint4*)(bbase + kc);
        b.u[1] = *(const uint4*)(bbase + kc + 16);
      } else {
        b.u[0] = make_uint4(0u, 0u, 0u, 0u);
        b.u[1] = make_uint4(0u, 0u, 0u, 0u);
      }
#pragma unroll
      for (int j = 0; j < 4; ++j) {
        VecBF a;
        const bf16* aj = abase + (size_t)j * 16 * 1152 + kc;
        a.u[0] = *(const uint4*)(aj);
        a.u[1] = *(const uint4*)(aj + 16);
        acc[j] = __builtin_amdgcn_wmma_f32_16x16x32_bf16(false, a.v, false, b.v,
                                                         (short)0, acc[j], false, false);
      }
    }
  }
#pragma unroll
  for (int j = 0; j < 4; ++j) {
#pragma unroll
    for (int r = 0; r < 8; ++r) {
      const int ch = c0 + j * 16 + r + half * 8;
      const size_t idx = (size_t)ch * HW + pix;
      yout[idx] = acc[j][r] + bias[ch] + res[idx];
    }
  }
}

// ---------------------------------------------------------------------------
// Instance norm over spatial dims, per channel. Optional transposed add:
//   x = src[c][i] + addT[i*128 + c];  dst[c][i] = (x - mean) * rsqrt(var+1e-5)
// one block (256 threads) per channel.
// ---------------------------------------------------------------------------
__global__ void k_inorm(const float* __restrict__ src, const float* __restrict__ addT,
                        float* __restrict__ dst) {
  const int c = blockIdx.x;
  const int t = threadIdx.x;
  __shared__ float s1[256], s2[256];
  float sum = 0.f, sq = 0.f;
  for (int i = t; i < HW; i += 256) {
    float x = src[(size_t)c * HW + i];
    if (addT) x += addT[(size_t)i * DCH + c];
    sum += x; sq += x * x;
  }
  s1[t] = sum; s2[t] = sq;
  __syncthreads();
  for (int s = 128; s > 0; s >>= 1) {
    if (t < s) { s1[t] += s1[t + s]; s2[t] += s2[t + s]; }
    __syncthreads();
  }
  const float mean = s1[0] * (1.0f / HW);
  const float var  = s2[0] * (1.0f / HW) - mean * mean;
  const float inv  = rsqrtf(var + 1e-5f);
  for (int i = t; i < HW; i += 256) {
    float x = src[(size_t)c * HW + i];
    if (addT) x += addT[(size_t)i * DCH + c];
    dst[(size_t)c * HW + i] = (x - mean) * inv;
  }
}

// NCHW fp32 -> channel-last bf16 [pix][c]
__global__ void k_chlast(const float* __restrict__ src, bf16* __restrict__ dst) {
  int i = blockIdx.x * blockDim.x + threadIdx.x;
  if (i >= HW * DCH) return;
  int pix = i >> 7, c = i & 127;
  dst[i] = (bf16)src[(size_t)c * HW + pix];
}

// (O,I,3,3) fp32 -> wpk[o][tap][cin] bf16 (K-major for WMMA A)
__global__ void k_pack_convw(const float* __restrict__ w, bf16* __restrict__ wpk) {
  int i = blockIdx.x * blockDim.x + threadIdx.x;
  if (i >= 128 * 1152) return;
  int o = i / 1152, r = i - o * 1152;
  int t = r >> 7, cin = r & 127;
  wpk[i] = (bf16)w[(size_t)(o * 128 + cin) * 9 + t];
}

__global__ void k_f32_to_bf16(const float* __restrict__ s, bf16* __restrict__ d, int n) {
  int i = blockIdx.x * blockDim.x + threadIdx.x;
  if (i < n) d[i] = (bf16)s[i];
}

// feat (1,N,C,Hl,Wl) fp32 -> dst[(cam*Hl+y)*Wl+x][c] fp32 (channel-last)
__global__ void k_transpose_feat(const float* __restrict__ f, float* __restrict__ dst,
                                 int Hl, int Wl) {
  int pos = blockIdx.x;           // (cam*Hl + y)*Wl + x
  int c = threadIdx.x;            // 128 threads
  int x = pos % Wl; int rest = pos / Wl;
  int y = rest % Hl; int cam = rest / Hl;
  dst[(size_t)pos * DCH + c] = f[(((size_t)cam * DCH + c) * Hl + y) * Wl + x];
}

// ---------------------------------------------------------------------------
// Per-query projections: offsets (sigmoid->scale->pts, height z), sample
// weights (softmax over L). One block (64 threads) per query.
// ---------------------------------------------------------------------------
__global__ void k_qproj(const float* __restrict__ q1, const float* __restrict__ bev_pos,
                        const float* __restrict__ off_w, const float* __restrict__ off_b,
                        const float* __restrict__ sw_w, const float* __restrict__ sw_b,
                        float* __restrict__ pts, float* __restrict__ sw,
                        float* __restrict__ height) {
  const int q = blockIdx.x;
  const int t = threadIdx.x;     // 64 threads
  __shared__ float qf[128];
  __shared__ float sraw[32];
  qf[t]      = q1[(size_t)t * HW + q];
  qf[t + 64] = q1[(size_t)(t + 64) * HW + q];
  __syncthreads();
  const float pcmin[3] = {-50.f, -50.f, -4.f};
  const float pcrng[3] = {100.f, 100.f, 8.f};
  if (t < 24) {
    float dot = off_b[t];
    for (int c = 0; c < 128; ++c) dot += qf[c] * off_w[t * 128 + c];
    float o = 1.f / (1.f + expf(-dot));
    int p = t / 3, dd = t - p * 3;
    float lim = (dd < 2) ? (0.25f + 1e-6f) : (4.f + 1e-6f);
    float val = o * 2.f * lim - lim;
    float ref = bev_pos[(size_t)q * 3 + dd] * pcrng[dd] + pcmin[dd];
    pts[((size_t)q * PPT + p) * 3 + dd] = ref + val;
    if (dd == 2) height[(size_t)p * HW + q] = val;
  }
  if (t >= 32) {
    int j = t - 32;
    float dot = sw_b[j];
    for (int c = 0; c < 128; ++c) dot += qf[c] * sw_w[j * 128 + c];
    sraw[j] = dot;
  }
  __syncthreads();
  if (t >= 32) {
    int j = t - 32;
    int p = j >> 2;
    float m = sraw[p * 4];
    for (int l = 1; l < 4; ++l) m = fmaxf(m, sraw[p * 4 + l]);
    float ssum = 0.f;
    for (int l = 0; l < 4; ++l) ssum += expf(sraw[p * 4 + l] - m);
    sw[(size_t)q * 32 + j] = expf(sraw[j] - m) / ssum;
  }
}

// ---------------------------------------------------------------------------
// Deformable 4D sampling + positional-encoding MLP epilogue.
// One block (128 threads = channels) per (query, point).
// Writes bf16 activation row A[q][p*128+c] for the mid-MLP GEMM.
// ---------------------------------------------------------------------------
__global__ void k_sample(const float* __restrict__ pts, const float* __restrict__ sw,
                         const float* __restrict__ featsT, const float* __restrict__ l2i,
                         const float* __restrict__ pe_w1, const float* __restrict__ pe_b1,
                         const float* __restrict__ pe_w2, const float* __restrict__ pe_b2,
                         bf16* __restrict__ Amid) {
  const int b = blockIdx.x;             // q*8 + p
  const int q = b >> 3, p = b & 7;
  const int c = threadIdx.x;            // 128 threads
  __shared__ float sm[96];
  __shared__ float hid[256];
  __shared__ float spt[3];
  __shared__ float ssw[4];
  if (c < 96) sm[c] = l2i[c];
  if (c < 3)  spt[c] = pts[((size_t)q * PPT + p) * 3 + c];
  if (c < 4)  ssw[c] = sw[(size_t)q * 32 + p * 4 + c];
  __syncthreads();
  const int    LH[4]   = {32, 16, 8, 4};
  const int    LW[4]   = {88, 44, 22, 11};
  const size_t LOFF[4] = {0, 2162688, 2703360, 2838528};
  const float X = spt[0], Y = spt[1], Z = spt[2];
  float acc = 0.f;
  for (int cam = 0; cam < NCAM; ++cam) {
    const float* M = sm + cam * 16;
    float cz = M[8] * X + M[9] * Y + M[10] * Z + M[11];
    if (cz <= 1e-5f) continue;                // ref: xy=-10000 -> all taps OOB
    float inv = 1.f / cz;
    float xn = (M[0] * X + M[1] * Y + M[2] * Z + M[3]) * inv * (1.f / 704.f);
    float yn = (M[4] * X + M[5] * Y + M[6] * Z + M[7]) * inv * (1.f / 256.f);
    for (int l = 0; l < NLVL; ++l) {
      float wl  = ssw[l];
      float pxf = xn * LW[l] - 0.5f;
      float pyf = yn * LH[l] - 0.5f;
      if (!(pxf > -1.f && pxf < (float)LW[l] && pyf > -1.f && pyf < (float)LH[l]))
        continue;
      float x0f = floorf(pxf), y0f = floorf(pyf);
      float wx = pxf - x0f, wy = pyf - y0f;
      int x0 = (int)x0f, y0 = (int)y0f;
      const float* base = featsT + LOFF[l] + (size_t)cam * LH[l] * LW[l] * DCH;
#pragma unroll
      for (int tap = 0; tap < 4; ++tap) {
        int xi = x0 + (tap & 1);
        int yi = y0 + (tap >> 1);
        if (xi < 0 || xi >= LW[l] || yi < 0 || yi >= LH[l]) continue;
        float wt = ((tap & 1) ? wx : 1.f - wx) * ((tap >> 1) ? wy : 1.f - wy);
        acc += wl * wt * base[((size_t)yi * LW[l] + xi) * DCH + c];
      }
    }
  }
  // positional encoding: relu(ref_n @ pe_w1 + b1) @ pe_w2 + b2
  float r0 = (X + 50.f) * 0.01f;
  float r1 = (Y + 50.f) * 0.01f;
  float r2 = (Z + 4.f)  * 0.125f;
  for (int j = c; j < 256; j += 128) {
    float h = pe_b1[j] + r0 * pe_w1[j] + r1 * pe_w1[256 + j] + r2 * pe_w1[512 + j];
    hid[j] = fmaxf(h, 0.f);
  }
  __syncthreads();
  float pe = pe_b2[c];
  for (int j = 0; j < 256; ++j) pe += hid[j] * pe_w2[j * 128 + c];
  Amid[(size_t)b * DCH + c] = (bf16)(acc + pe);
}

// ---------------------------------------------------------------------------
extern "C" void kernel_launch(void* const* d_in, const int* in_sizes, int n_in,
                              void* d_out, int out_size, void* d_ws, size_t ws_size,
                              hipStream_t stream) {
  (void)in_sizes; (void)n_in; (void)out_size; (void)ws_size;
  const float* feat0  = (const float*)d_in[0];
  const float* feat1  = (const float*)d_in[1];
  const float* feat2  = (const float*)d_in[2];
  const float* feat3  = (const float*)d_in[3];
  const float* l2i    = (const float*)d_in[4];
  const float* bevq   = (const float*)d_in[5];
  const float* bevp   = (const float*)d_in[6];
  const float* in_w   = (const float*)d_in[7];
  const float* in_b   = (const float*)d_in[8];
  const float* off_w  = (const float*)d_in[9];
  const float* off_b  = (const float*)d_in[10];
  const float* sw_w   = (const float*)d_in[11];
  const float* sw_b   = (const float*)d_in[12];
  const float* pe_w1  = (const float*)d_in[13];
  const float* pe_b1  = (const float*)d_in[14];
  const float* pe_w2  = (const float*)d_in[15];
  const float* pe_b2  = (const float*)d_in[16];
  const float* mid_w1 = (const float*)d_in[17];
  const float* mid_b1 = (const float*)d_in[18];
  const float* mid_w2 = (const float*)d_in[19];
  const float* mid_b2 = (const float*)d_in[20];
  const float* mid_w3 = (const float*)d_in[21];
  const float* mid_b3 = (const float*)d_in[22];
  const float* out_w  = (const float*)d_in[23];
  const float* out_b  = (const float*)d_in[24];

  char* wsb = (char*)d_ws;
  size_t off = 0;
  auto alloc = [&](size_t bytes) -> void* {
    off = (off + 255) & ~(size_t)255;
    void* p = wsb + off;
    off += bytes;
    return p;
  };
  float* q1     = (float*)alloc((size_t)HW * DCH * 4);         // normed query
  float* yc     = (float*)alloc((size_t)HW * DCH * 4);         // conv scratch
  float* featsT = (float*)alloc((size_t)2872320 * 4);          // channel-last feats
  float* pts    = (float*)alloc((size_t)HW * PPT * 3 * 4);
  float* swb    = (float*)alloc((size_t)HW * 32 * 4);
  bf16*  Amid   = (bf16*)alloc((size_t)HW * 1024 * 2);         // sampled+pe, bf16
  bf16*  m1     = (bf16*)alloc((size_t)HW * 512 * 2);
  float* m3     = (float*)alloc((size_t)HW * 128 * 4);
  bf16*  xT     = (bf16*)alloc((size_t)HW * DCH * 2);          // channel-last bf16
  bf16*  wpk_in  = (bf16*)alloc((size_t)147456 * 2);
  bf16*  wpk_out = (bf16*)alloc((size_t)147456 * 2);
  bf16*  w1bf   = (bf16*)alloc((size_t)512 * 1024 * 2);
  bf16*  w2bf   = (bf16*)alloc((size_t)512 * 512 * 2);
  bf16*  w3bf   = (bf16*)alloc((size_t)128 * 512 * 2);
  // Aliases into the (dead-after-GEMM1) Amid region:
  bf16*  m2 = Amid;                                            // 16 MB
  float* q2 = (float*)((char*)Amid + (size_t)HW * 512 * 2);    // 8 MB, 2nd half

  float* qout = (float*)d_out;                 // (1,128,128,128)
  float* hout = qout + (size_t)HW * DCH;       // (1,1,8,128,128)

  // ---- weight prep (bf16 packing) ----
  k_pack_convw<<<(147456 + 255) / 256, 256, 0, stream>>>(in_w, wpk_in);
  k_pack_convw<<<(147456 + 255) / 256, 256, 0, stream>>>(out_w, wpk_out);
  k_f32_to_bf16<<<(524288 + 255) / 256, 256, 0, stream>>>(mid_w1, w1bf, 524288);
  k_f32_to_bf16<<<(262144 + 255) / 256, 256, 0, stream>>>(mid_w2, w2bf, 262144);
  k_f32_to_bf16<<<(65536 + 255) / 256, 256, 0, stream>>>(mid_w3, w3bf, 65536);

  // ---- features -> channel-last (L2-resident gather layout) ----
  k_transpose_feat<<<6 * 32 * 88, 128, 0, stream>>>(feat0, featsT, 32, 88);
  k_transpose_feat<<<6 * 16 * 44, 128, 0, stream>>>(feat1, featsT + 2162688, 16, 44);
  k_transpose_feat<<<6 * 8 * 22, 128, 0, stream>>>(feat2, featsT + 2703360, 8, 22);
  k_transpose_feat<<<6 * 4 * 11, 128, 0, stream>>>(feat3, featsT + 2838528, 4, 11);

  // ---- q = inorm(bev_query + conv3x3(bev_query)) ----
  k_chlast<<<(HW * DCH + 255) / 256, 256, 0, stream>>>(bevq, xT);
  k_conv3x3_wmma<<<dim3(1024, 2), 32, 0, stream>>>(xT, wpk_in, in_b, bevq, yc);
  k_inorm<<<128, 256, 0, stream>>>(yc, nullptr, q1);

  // ---- offsets / sample weights / points / height ----
  k_qproj<<<HW, 64, 0, stream>>>(q1, bevp, off_w, off_b, sw_w, sw_b, pts, swb, hout);

  // ---- deformable sampling + PE MLP -> bf16 activation matrix ----
  k_sample<<<HW * PPT, 128, 0, stream>>>(pts, swb, featsT, l2i,
                                         pe_w1, pe_b1, pe_w2, pe_b2, Amid);

  // ---- mid MLP: 1024 -> 512 (gelu) -> 512 (gelu) -> 128 ----
  k_gemm_wmma<<<dim3(1024, 8), 32, 0, stream>>>(Amid, w1bf, mid_b1, m1, 1024, 512, 1, 1);
  k_gemm_wmma<<<dim3(1024, 8), 32, 0, stream>>>(m1, w2bf, mid_b2, m2, 512, 512, 1, 1);
  k_gemm_wmma<<<dim3(1024, 2), 32, 0, stream>>>(m2, w3bf, mid_b3, m3, 512, 128, 0, 0);

  // ---- q = inorm(q + m) ----
  k_inorm<<<128, 256, 0, stream>>>(q1, m3, q2);

  // ---- q = inorm(q + conv3x3(q)) -> final output ----
  k_chlast<<<(HW * DCH + 255) / 256, 256, 0, stream>>>(q2, xT);
  k_conv3x3_wmma<<<dim3(1024, 2), 32, 0, stream>>>(xT, wpk_out, out_b, q2, yc);
  k_inorm<<<128, 256, 0, stream>>>(yc, nullptr, qout);
}
